// TensorProduct_81673098100901
// MI455X (gfx1250) — compile-verified, matrix operand order checked by reading, not run on hardware
//
#include <hip/hip_runtime.h>

typedef __attribute__((ext_vector_type(16))) _Float16 v16h;
typedef __attribute__((ext_vector_type(8)))  float    v8f;

#define MULC   64
#define O0C    320
#define O1MC   128
#define O2MC   64
#define NROWS  100000
#define NTILES (NROWS / 16)                // 6250
#define IN1S   (4 * MULC)                  // 256 floats per input_1 row
#define OUTS   (O0C + 3 * O1MC + 3 * O2MC) // 896 floats per output row

// packed weight fragments: each fragment = 32 lanes x 16 halves = 512 halves (1 KB)
#define FR_WA0    0     // 40 frags (20 ntiles x 2 ksteps), 64->320
#define FR_WB1O0  40    // 40 frags
#define FR_WA1O1  80    // 16 frags, 64->128
#define FR_WB0O1  96    // 16 frags
#define FR_WB1O2  112   // 8 frags, 64->64
#define NFRAGS    120
#define PACKED_BYTES (NFRAGS * 512 * 2)    // 122880 B

// half-index -> K mapping for 16-bit A (16x32) and B (32x16) fragments (ISA 7.12.2)
__device__ __forceinline__ int kmap(int j, int hi) {
  return (hi ? 8 : 0) + (j < 8 ? j : j + 8);
}

__device__ __forceinline__ v8f wmma_f16(v16h a, v16h b, v8f c) {
  return __builtin_amdgcn_wmma_f32_16x16x32_f16(false, a, false, b, (short)0, c,
                                                false, false);
}

// ---------------- weight pre-pack: f32 (K=64, Ncols) -> f16 WMMA-B fragments ----
__global__ void pack_weights_kernel(const float* __restrict__ Wa0,
                                    const float* __restrict__ Wb1o0,
                                    const float* __restrict__ Wa1o1,
                                    const float* __restrict__ Wb0o1,
                                    const float* __restrict__ Wb1o2,
                                    _Float16* __restrict__ packed) {
  int t = blockIdx.x * blockDim.x + threadIdx.x;   // one thread per (frag, lane)
  if (t >= NFRAGS * 32) return;
  int frag = t >> 5, lane = t & 31;
  const float* W; int ncols, f;
  if (frag < FR_WB1O0)      { W = Wa0;   ncols = O0C;  f = frag; }
  else if (frag < FR_WB0O1) {
    if (frag < FR_WA1O1)    { W = Wb1o0; ncols = O0C;  f = frag - FR_WB1O0; }
    else                    { W = Wa1o1; ncols = O1MC; f = frag - FR_WA1O1; }
  }
  else if (frag < FR_WB1O2) { W = Wb0o1; ncols = O1MC; f = frag - FR_WB0O1; }
  else                      { W = Wb1o2; ncols = O2MC; f = frag - FR_WB1O2; }
  int ntile = f >> 1, kstep = f & 1;
  int hi = lane >> 4, col = ntile * 16 + (lane & 15);
  _Float16* out = packed + ((long)frag * 32 + lane) * 16;
#pragma unroll
  for (int j = 0; j < 16; ++j) {
    int k = kstep * 32 + kmap(j, hi);
    out[j] = (_Float16)W[k * ncols + col];
  }
}

// ---------------- main tensor-product kernel: 1 wave per 16-row tile ----------
__global__ void __launch_bounds__(256)
tp_kernel(const float* __restrict__ in1, const float* __restrict__ in2,
          const float* __restrict__ bias, const _Float16* __restrict__ packed,
          float* __restrict__ out) {
  extern __shared__ __attribute__((aligned(32))) _Float16 smW[];
  // cooperative stage of packed weights into LDS (7680 x uint4)
  {
    const uint4* src = (const uint4*)packed;
    uint4* dst = (uint4*)smW;
    for (int i = threadIdx.x; i < PACKED_BYTES / 16; i += 256) dst[i] = src[i];
  }
  __syncthreads();

  int wave = threadIdx.x >> 5;
  int lane = threadIdx.x & 31;
  int tile = blockIdx.x * 8 + wave;
  if (tile >= NTILES) return;

  int row0 = tile * 16;
  int m = lane & 15, hi = lane >> 4;
  const v16h* B = (const v16h*)smW;                 // fragment f, lane l -> B[f*32+l]

  const float* r1 = in1 + (long)(row0 + m) * IN1S;  // this lane's A-row
  const float* r2 = in2 + (long)(row0 + m) * 4;
  float s0v = r2[0], v1x = r2[1], v1y = r2[2], v1z = r2[3];

  const float scale  = 0.08838834764831845f;        // 1/sqrt(128) == PW0 == PW1/sqrt3 == PW2/sqrt6
  const float inv_s3 = 0.5773502691896258f;

  // per-lane v1 for the 8 output rows this lane's D-fragment elements belong to
  float v1r[8][3];
#pragma unroll
  for (int i = 0; i < 8; ++i) {
    const float* q = in2 + (long)(row0 + i + hi * 8) * 4;
    v1r[i][0] = q[1]; v1r[i][1] = q[2]; v1r[i][2] = q[3];
  }

  // ---------------- phase 1: o0 = scale*(A_s0@Wa0 + A_d@Wb1o0) + bias ----------
  {
    v16h a_s0[2], a_d[2];
#pragma unroll
    for (int ks = 0; ks < 2; ++ks) {
#pragma unroll
      for (int j = 0; j < 16; ++j) {
        int u = ks * 32 + kmap(j, hi);
        float sa = r1[u];
        const float* vb = r1 + MULC + 3 * u;
        float dot = vb[0] * v1x + vb[1] * v1y + vb[2] * v1z;
        a_s0[ks][j] = (_Float16)(sa * s0v);
        a_d[ks][j]  = (_Float16)(dot * inv_s3);
      }
    }
    // software-pipelined B-fragment prefetch (1 tile lookahead)
    v16h c0 = B[(FR_WA0 + 0) * 32 + lane];
    v16h c1 = B[(FR_WA0 + 1) * 32 + lane];
    v16h c2 = B[(FR_WB1O0 + 0) * 32 + lane];
    v16h c3 = B[(FR_WB1O0 + 1) * 32 + lane];
#pragma unroll
    for (int nt = 0; nt < 20; ++nt) {
      int nn = (nt + 1 < 20) ? nt + 1 : nt;   // compile-time under unroll
      v16h n0 = B[(FR_WA0 + nn * 2 + 0) * 32 + lane];
      v16h n1 = B[(FR_WA0 + nn * 2 + 1) * 32 + lane];
      v16h n2 = B[(FR_WB1O0 + nn * 2 + 0) * 32 + lane];
      v16h n3 = B[(FR_WB1O0 + nn * 2 + 1) * 32 + lane];
      v8f acc = {};
      acc = wmma_f16(a_s0[0], c0, acc);
      acc = wmma_f16(a_s0[1], c1, acc);
      acc = wmma_f16(a_d[0],  c2, acc);
      acc = wmma_f16(a_d[1],  c3, acc);
      float bv = bias[nt * 16 + m];
#pragma unroll
      for (int i = 0; i < 8; ++i) {
        long row = row0 + i + hi * 8;
        out[row * OUTS + nt * 16 + m] = scale * acc[i] + bv;
      }
      c0 = n0; c1 = n1; c2 = n2; c3 = n3;
    }
  }

  // ---------------- phase 2: o1 = scale*( (s_a@Wa1o1)*v1 + (s0*v_b)@Wb0o1 ) ----
  {
    v16h a_sa[2], a_vb[3][2];
#pragma unroll
    for (int ks = 0; ks < 2; ++ks) {
#pragma unroll
      for (int j = 0; j < 16; ++j) {
        int u = ks * 32 + kmap(j, hi);
        const float* vb = r1 + MULC + 3 * u;
        a_sa[ks][j]    = (_Float16)r1[u];
        a_vb[0][ks][j] = (_Float16)(vb[0] * s0v);
        a_vb[1][ks][j] = (_Float16)(vb[1] * s0v);
        a_vb[2][ks][j] = (_Float16)(vb[2] * s0v);
      }
    }
    v16h c0 = B[(FR_WA1O1 + 0) * 32 + lane];
    v16h c1 = B[(FR_WA1O1 + 1) * 32 + lane];
    v16h c2 = B[(FR_WB0O1 + 0) * 32 + lane];
    v16h c3 = B[(FR_WB0O1 + 1) * 32 + lane];
#pragma unroll
    for (int nt = 0; nt < 8; ++nt) {
      int nn = (nt + 1 < 8) ? nt + 1 : nt;
      v16h n0 = B[(FR_WA1O1 + nn * 2 + 0) * 32 + lane];
      v16h n1 = B[(FR_WA1O1 + nn * 2 + 1) * 32 + lane];
      v16h n2 = B[(FR_WB0O1 + nn * 2 + 0) * 32 + lane];
      v16h n3 = B[(FR_WB0O1 + nn * 2 + 1) * 32 + lane];
      v8f ta = {};
      ta = wmma_f16(a_sa[0], c0, ta);
      ta = wmma_f16(a_sa[1], c1, ta);
#pragma unroll
      for (int k = 0; k < 3; ++k) {
        v8f tb = {};
        tb = wmma_f16(a_vb[k][0], c2, tb);
        tb = wmma_f16(a_vb[k][1], c3, tb);
#pragma unroll
        for (int i = 0; i < 8; ++i) {
          long row = row0 + i + hi * 8;
          out[row * OUTS + O0C + (nt * 16 + m) * 3 + k] =
              scale * (ta[i] * v1r[i][k] + tb[i]);
        }
      }
      c0 = n0; c1 = n1; c2 = n2; c3 = n3;
    }
  }

  // ---------------- phase 3: o2 = scale*( cross(v_b, v1)@Wb1o2 ) ---------------
  {
    v16h a_cr[3][2];
#pragma unroll
    for (int ks = 0; ks < 2; ++ks) {
#pragma unroll
      for (int j = 0; j < 16; ++j) {
        int u = ks * 32 + kmap(j, hi);
        const float* vb = r1 + MULC + 3 * u;
        a_cr[0][ks][j] = (_Float16)(vb[1] * v1z - vb[2] * v1y);
        a_cr[1][ks][j] = (_Float16)(vb[2] * v1x - vb[0] * v1z);
        a_cr[2][ks][j] = (_Float16)(vb[0] * v1y - vb[1] * v1x);
      }
    }
    v16h c0 = B[(FR_WB1O2 + 0) * 32 + lane];
    v16h c1 = B[(FR_WB1O2 + 1) * 32 + lane];
#pragma unroll
    for (int nt = 0; nt < 4; ++nt) {
      int nn = (nt + 1 < 4) ? nt + 1 : nt;
      v16h n0 = B[(FR_WB1O2 + nn * 2 + 0) * 32 + lane];
      v16h n1 = B[(FR_WB1O2 + nn * 2 + 1) * 32 + lane];
#pragma unroll
      for (int k = 0; k < 3; ++k) {
        v8f c = {};
        c = wmma_f16(a_cr[k][0], c0, c);
        c = wmma_f16(a_cr[k][1], c1, c);
#pragma unroll
        for (int i = 0; i < 8; ++i) {
          long row = row0 + i + hi * 8;
          out[row * OUTS + O0C + 3 * O1MC + (nt * 16 + m) * 3 + k] = scale * c[i];
        }
      }
      c0 = n0; c1 = n1;
    }
  }
}

extern "C" void kernel_launch(void* const* d_in, const int* in_sizes, int n_in,
                              void* d_out, int out_size, void* d_ws, size_t ws_size,
                              hipStream_t stream) {
  const float* in1   = (const float*)d_in[0];
  const float* in2   = (const float*)d_in[1];
  const float* Wa0   = (const float*)d_in[2];
  const float* Wb1o0 = (const float*)d_in[3];
  const float* Wa1o1 = (const float*)d_in[4];
  const float* Wb0o1 = (const float*)d_in[5];
  const float* Wb1o2 = (const float*)d_in[6];
  const float* bias  = (const float*)d_in[7];
  _Float16* packed = (_Float16*)d_ws;
  float* out = (float*)d_out;

  pack_weights_kernel<<<(NFRAGS * 32 + 255) / 256, 256, 0, stream>>>(
      Wa0, Wb1o0, Wa1o1, Wb0o1, Wb1o2, packed);

  int nblocks = (NTILES + 7) / 8;   // 8 waves (tiles) per 256-thread block
  tp_kernel<<<nblocks, 256, PACKED_BYTES, stream>>>(in1, in2, bias, packed, out);
}
